// PatchTST_backbone_attn_weight_global_37666863186479
// MI455X (gfx1250) — compile-verified, matrix operand order checked by reading, not run on hardware
//
#include <hip/hip_runtime.h>

// ---------------------------------------------------------------------------
// Problem constants (from reference): B=256, S=128, D=512, H=8, F=2048, NL=3
// ---------------------------------------------------------------------------
constexpr int CB = 256;
constexpr int CS = 128;
constexpr int CD = 512;
constexpr int CH = 8;
constexpr int CF = 2048;
constexpr int CNL = 3;
constexpr int CDH = 64;                 // D / H
constexpr float CSCALE = 0.125f;        // DH^-0.5
constexpr float CLN_EPS = 1e-5f;

typedef __attribute__((ext_vector_type(16))) __bf16 v16bf;
typedef __attribute__((ext_vector_type(8)))  float  v8f;

// Build a 16-element bf16 WMMA fragment from two contiguous 16-byte chunks.
// ISA layout (16-bit A/B 16x32): lane<16 -> K base 0, lane>=16 -> K base 8;
// vector elems 0..7 = K[base..base+7], elems 8..15 = K[base+16..base+23].
__device__ __forceinline__ v16bf load_frag(const __bf16* c0, const __bf16* c1) {
    union { uint4 u[2]; v16bf v; } f;
    f.u[0] = *reinterpret_cast<const uint4*>(c0);
    f.u[1] = *reinterpret_cast<const uint4*>(c1);
    return f.v;
}

__device__ __forceinline__ v8f wmma_bf16(v16bf a, v16bf b, v8f c) {
    return __builtin_amdgcn_wmma_f32_16x16x32_bf16(false, a, false, b,
                                                   (short)0, c, false, false);
}

// CDNA5 async global->LDS copy (16B per lane). VDST VGPR carries the per-lane
// LDS byte offset (= low 32 bits of the generic shared address, per the flat
// aperture mapping); VADDR is the 64-bit global address.
// Tracked by ASYNCcnt (in-order completion), so a partial wait releases the
// oldest batch while a prefetch batch stays in flight.
__device__ __forceinline__ void async_copy_b128(void* lds_dst, const void* gsrc) {
    unsigned loff = (unsigned)(size_t)lds_dst;
    unsigned long long ga = (unsigned long long)gsrc;
    asm volatile("global_load_async_to_lds_b128 %0, %1, off"
                 :: "v"(loff), "v"(ga)
                 : "memory");
}
template <int N>
__device__ __forceinline__ void wait_async_le() {
    asm volatile("s_wait_asynccnt %0" :: "i"(N) : "memory");
}

// ---------------------------------------------------------------------------
// Generic bf16 GEMM:  out[M,N] = A[M,K] @ W^T[N,K] + bias  (+ epilogue)
//   Wt is PRE-TRANSPOSED [N][K] so both LDS stages are contiguous 16B copies.
// mode 0: fp32 out; 1: bf16 out; 2: exact GELU -> bf16; 3: bf16 out in
//         attention layout [b,h,dh,s] (used for the V projection).
// BM=BN=128, BK=32, double-buffered async staging. 256 threads = 8 waves as
// 4(M)x2(N); wave strip 32x64: 2 A-frags x 4 B-frags -> 8 WMMAs per K-step.
// ---------------------------------------------------------------------------
__global__ __launch_bounds__(256) void gemm_bf16(
    const __bf16* __restrict__ A, const __bf16* __restrict__ Wt,
    const float* __restrict__ bias, float* __restrict__ outF,
    __bf16* __restrict__ outB, int M, int N, int K, int mode)
{
    __shared__ __bf16 As[2][128][40];   // [m][k], 32 + 8 pad (16B-aligned rows)
    __shared__ __bf16 Bs[2][128][40];   // [n][k]

    const int tid  = threadIdx.x;
    const int lane = tid & 31;
    const int wave = tid >> 5;
    const bool hi  = (lane & 16) != 0;
    const int ln   = lane & 15;
    const int kbo  = hi ? 8 : 0;
    const int m0   = (wave >> 1) * 32;   // M sub-strip of 32 rows
    const int n0w  = (wave & 1) * 64;    // N sub-strip of 64 cols

    const long row0 = (long)blockIdx.y * 128;
    const int  col0 = blockIdx.x * 128;

    const v8f vz = {0.f,0.f,0.f,0.f,0.f,0.f,0.f,0.f};
    v8f acc[2][4] = {{vz, vz, vz, vz}, {vz, vz, vz, vz}};

    const int srow = tid >> 1, shalf = (tid & 1) * 16;  // staging: 16 elems/thread
    const __bf16* agp = A  + (row0 + srow) * (long)K + shalf;
    const __bf16* bgp = Wt + (long)(col0 + srow) * K + shalf;

    auto stage = [&](int p, int kk) {
        async_copy_b128(&As[p][srow][shalf],     agp + kk);
        async_copy_b128(&As[p][srow][shalf + 8], agp + kk + 8);
        async_copy_b128(&Bs[p][srow][shalf],     bgp + kk);
        async_copy_b128(&Bs[p][srow][shalf + 8], bgp + kk + 8);
    };

    stage(0, 0);
    int p = 0;
    for (int kk = 0; kk < K; kk += 32) {
        const bool more = (kk + 32) < K;
        if (more) { stage(p ^ 1, kk + 32); wait_async_le<4>(); }
        else      { wait_async_le<0>(); }
        __syncthreads();

        v16bf a0 = load_frag(&As[p][m0 + ln][kbo],      &As[p][m0 + ln][kbo + 16]);
        v16bf a1 = load_frag(&As[p][m0 + 16 + ln][kbo], &As[p][m0 + 16 + ln][kbo + 16]);
        #pragma unroll
        for (int nt = 0; nt < 4; ++nt) {
            v16bf bf = load_frag(&Bs[p][n0w + nt*16 + ln][kbo],
                                 &Bs[p][n0w + nt*16 + ln][kbo + 16]);
            acc[0][nt] = wmma_bf16(a0, bf, acc[0][nt]);
            acc[1][nt] = wmma_bf16(a1, bf, acc[1][nt]);
        }
        __syncthreads();
        p ^= 1;
    }

    // C/D layout: VGPR r -> M = r (lanes 0..15) or r+8 (lanes 16..31); N = lane&15
    // Mode dispatch hoisted outside the unrolled store loops.
    if (mode == 0) {
        #pragma unroll
        for (int mi = 0; mi < 2; ++mi)
        #pragma unroll
        for (int nt = 0; nt < 4; ++nt) {
            const int colG = col0 + n0w + nt*16 + ln;
            const float bv = bias[colG];
            #pragma unroll
            for (int r = 0; r < 8; ++r) {
                const long rowG = row0 + m0 + mi*16 + (hi ? r + 8 : r);
                outF[rowG * N + colG] = acc[mi][nt][r] + bv;
            }
        }
    } else if (mode == 1) {
        #pragma unroll
        for (int mi = 0; mi < 2; ++mi)
        #pragma unroll
        for (int nt = 0; nt < 4; ++nt) {
            const int colG = col0 + n0w + nt*16 + ln;
            const float bv = bias[colG];
            #pragma unroll
            for (int r = 0; r < 8; ++r) {
                const long rowG = row0 + m0 + mi*16 + (hi ? r + 8 : r);
                outB[rowG * N + colG] = (__bf16)(acc[mi][nt][r] + bv);
            }
        }
    } else if (mode == 2) {
        #pragma unroll
        for (int mi = 0; mi < 2; ++mi)
        #pragma unroll
        for (int nt = 0; nt < 4; ++nt) {
            const int colG = col0 + n0w + nt*16 + ln;
            const float bv = bias[colG];
            #pragma unroll
            for (int r = 0; r < 8; ++r) {
                const long rowG = row0 + m0 + mi*16 + (hi ? r + 8 : r);
                float v = acc[mi][nt][r] + bv;
                v = 0.5f * v * (1.0f + erff(v * 0.70710678118654752f));
                outB[rowG * N + colG] = (__bf16)v;
            }
        }
    } else {  // mode 3: [b,h,dh,s] for the V projection
        #pragma unroll
        for (int mi = 0; mi < 2; ++mi)
        #pragma unroll
        for (int nt = 0; nt < 4; ++nt) {
            const int colG = col0 + n0w + nt*16 + ln;
            const float bv = bias[colG];
            const int h_ = colG >> 6, dh_ = colG & 63;
            #pragma unroll
            for (int r = 0; r < 8; ++r) {
                const long rowG = row0 + m0 + mi*16 + (hi ? r + 8 : r);
                const long b_ = rowG >> 7;
                const int  s_ = (int)(rowG & 127);
                outB[((b_ * CH + h_) * CDH + dh_) * CS + s_] =
                    (__bf16)(acc[mi][nt][r] + bv);
            }
        }
    }
}

// ---------------------------------------------------------------------------
// Attention scores: per (b,h) block computes S[128,128] = Q @ K^T,
// applies mask_col * scale, adds prev scores (RealFormer), stores to prev.
// K rows are naturally the [n][k] B-operand (k = dh contiguous).
// ---------------------------------------------------------------------------
__global__ __launch_bounds__(256) void attn_scores(
    const __bf16* __restrict__ qb, const __bf16* __restrict__ kb,
    const float* __restrict__ mask_col, float* __restrict__ prev, int has_prev)
{
    __shared__ __bf16 Qs[128][72];   // DH=64 + 8 pad
    __shared__ __bf16 Ks[128][72];

    const int bh = blockIdx.x;
    const int b = bh / CH, h = bh % CH;
    const int tid = threadIdx.x, lane = tid & 31, wave = tid >> 5;
    const bool hi = (lane & 16) != 0;
    const int ln = lane & 15, kbo = hi ? 8 : 0;

    const long base = ((long)b * CS) * CD + h * CDH;   // [b, s, h*DH]

    {   // stage Q and K tiles [128, 64] via async copies (4 x 16B each)
        const int row = tid >> 1, half = (tid & 1) * 32;
        #pragma unroll
        for (int j = 0; j < 4; ++j) {
            async_copy_b128(&Qs[row][half + j*8], qb + base + (long)row * CD + half + j*8);
            async_copy_b128(&Ks[row][half + j*8], kb + base + (long)row * CD + half + j*8);
        }
        wait_async_le<0>();
    }
    __syncthreads();

    const v8f vz = {0.f,0.f,0.f,0.f,0.f,0.f,0.f,0.f};
    v8f acc[8] = {vz, vz, vz, vz, vz, vz, vz, vz};

    #pragma unroll
    for (int kk = 0; kk < CDH; kk += 32) {
        v16bf af = load_frag(&Qs[wave*16 + ln][kk + kbo], &Qs[wave*16 + ln][kk + kbo + 16]);
        #pragma unroll
        for (int nt = 0; nt < 8; ++nt) {
            v16bf bf = load_frag(&Ks[nt*16 + ln][kk + kbo], &Ks[nt*16 + ln][kk + kbo + 16]);
            acc[nt] = wmma_bf16(af, bf, acc[nt]);
        }
    }

    const long pbase = (long)bh * CS * CS;
    if (has_prev) {
        #pragma unroll
        for (int nt = 0; nt < 8; ++nt) {
            const int col = nt*16 + ln;
            const float mc = mask_col[col] * CSCALE;
            #pragma unroll
            for (int r = 0; r < 8; ++r) {
                const int  row = wave*16 + (hi ? r + 8 : r);
                const long idx = pbase + (long)row * CS + col;
                prev[idx] = acc[nt][r] * mc + prev[idx];
            }
        }
    } else {
        #pragma unroll
        for (int nt = 0; nt < 8; ++nt) {
            const int col = nt*16 + ln;
            const float mc = mask_col[col] * CSCALE;
            #pragma unroll
            for (int r = 0; r < 8; ++r) {
                const int  row = wave*16 + (hi ? r + 8 : r);
                prev[pbase + (long)row * CS + col] = acc[nt][r] * mc;
            }
        }
    }
}

// ---------------------------------------------------------------------------
// Row softmax over S=128: one wave per row, 4 values per lane, shfl reduce.
// ---------------------------------------------------------------------------
__global__ __launch_bounds__(256) void attn_softmax(
    const float* __restrict__ sc, __bf16* __restrict__ pb)
{
    const int wave = threadIdx.x >> 5, lane = threadIdx.x & 31;
    const long row = (long)blockIdx.x * 8 + wave;
    const float4 v4 = *reinterpret_cast<const float4*>(sc + row * CS + lane * 4);

    float m = fmaxf(fmaxf(v4.x, v4.y), fmaxf(v4.z, v4.w));
    #pragma unroll
    for (int o = 16; o > 0; o >>= 1) m = fmaxf(m, __shfl_xor(m, o, 32));

    const float e0 = __expf(v4.x - m), e1 = __expf(v4.y - m);
    const float e2 = __expf(v4.z - m), e3 = __expf(v4.w - m);
    float s = e0 + e1 + e2 + e3;
    #pragma unroll
    for (int o = 16; o > 0; o >>= 1) s += __shfl_xor(s, o, 32);
    const float inv = 1.0f / s;

    union { __bf16 p[4]; uint2 u; } out;
    out.p[0] = (__bf16)(e0 * inv); out.p[1] = (__bf16)(e1 * inv);
    out.p[2] = (__bf16)(e2 * inv); out.p[3] = (__bf16)(e3 * inv);
    *reinterpret_cast<uint2*>(pb + row * CS + lane * 4) = out.u;
}

// ---------------------------------------------------------------------------
// out[b,s,h*DH+dh] = probs[128,128] @ V^T  per (b,h).
// vt is the V projection stored as [b,h,dh,s] (GEMM mode 3), so the B-operand
// tile Bs[n=dh][k=s] stages with contiguous 16B copies. Double-buffered.
// ---------------------------------------------------------------------------
__global__ __launch_bounds__(256) void attn_v(
    const __bf16* __restrict__ pb, const __bf16* __restrict__ vt,
    __bf16* __restrict__ ob)
{
    __shared__ __bf16 Ps[2][128][40];
    __shared__ __bf16 Vs[2][64][40];    // [dh][s]

    const int bh = blockIdx.x;
    const int b = bh / CH, h = bh % CH;
    const int tid = threadIdx.x, lane = tid & 31, wave = tid >> 5;
    const bool hi = (lane & 16) != 0;
    const int ln = lane & 15, kbo = hi ? 8 : 0;

    const long pbase = (long)bh * CS * CS;
    const long vtb   = (long)bh * CDH * CS;
    const long obase = ((long)b * CS) * CD + h * CDH;

    const v8f vz = {0.f,0.f,0.f,0.f,0.f,0.f,0.f,0.f};
    v8f acc[4] = {vz, vz, vz, vz};

    const int prow = tid >> 1, phalf = (tid & 1) * 16;
    const int vn   = tid >> 2, vkoff = (tid & 3) * 8;
    const __bf16* pgp = pb + pbase + (long)prow * CS + phalf;
    const __bf16* vgp = vt + vtb + (long)vn * CS + vkoff;

    auto stage = [&](int pg, int kk) {
        async_copy_b128(&Ps[pg][prow][phalf],     pgp + kk);
        async_copy_b128(&Ps[pg][prow][phalf + 8], pgp + kk + 8);
        async_copy_b128(&Vs[pg][vn][vkoff],       vgp + kk);
    };

    stage(0, 0);
    int pg = 0;
    for (int kk = 0; kk < CS; kk += 32) {
        const bool more = (kk + 32) < CS;
        if (more) { stage(pg ^ 1, kk + 32); wait_async_le<3>(); }
        else      { wait_async_le<0>(); }
        __syncthreads();

        v16bf af = load_frag(&Ps[pg][wave*16 + ln][kbo], &Ps[pg][wave*16 + ln][kbo + 16]);
        #pragma unroll
        for (int nt = 0; nt < 4; ++nt) {
            v16bf bf = load_frag(&Vs[pg][nt*16 + ln][kbo], &Vs[pg][nt*16 + ln][kbo + 16]);
            acc[nt] = wmma_bf16(af, bf, acc[nt]);
        }
        __syncthreads();
        pg ^= 1;
    }

    #pragma unroll
    for (int nt = 0; nt < 4; ++nt) {
        const int col = nt*16 + ln;
        #pragma unroll
        for (int r = 0; r < 8; ++r) {
            const int row = wave*16 + (hi ? r + 8 : r);
            ob[obase + (long)row * CD + col] = (__bf16)acc[nt][r];
        }
    }
}

// ---------------------------------------------------------------------------
// x = LayerNorm(x + t) * g + b  (fp32, D=512), also writes bf16 mirror.
// 128 threads/block, 4 elems/thread, one block per row.
// ---------------------------------------------------------------------------
__global__ __launch_bounds__(128) void resid_ln(
    float* __restrict__ x, const float* __restrict__ t,
    const float* __restrict__ g, const float* __restrict__ bta,
    __bf16* __restrict__ xb)
{
    __shared__ float redS[4], redQ[4];
    const long row = blockIdx.x;
    const int tid = threadIdx.x, lane = tid & 31, wave = tid >> 5;

    const float4 xv = *reinterpret_cast<const float4*>(x + row * CD + tid * 4);
    const float4 tv = *reinterpret_cast<const float4*>(t + row * CD + tid * 4);
    float v[4] = {xv.x + tv.x, xv.y + tv.y, xv.z + tv.z, xv.w + tv.w};

    float s  = v[0] + v[1] + v[2] + v[3];
    float sq = v[0]*v[0] + v[1]*v[1] + v[2]*v[2] + v[3]*v[3];
    #pragma unroll
    for (int o = 16; o > 0; o >>= 1) {
        s  += __shfl_xor(s,  o, 32);
        sq += __shfl_xor(sq, o, 32);
    }
    if (lane == 0) { redS[wave] = s; redQ[wave] = sq; }
    __syncthreads();
    const float S  = redS[0] + redS[1] + redS[2] + redS[3];
    const float Q  = redQ[0] + redQ[1] + redQ[2] + redQ[3];
    const float mu = S * (1.0f / CD);
    const float var = Q * (1.0f / CD) - mu * mu;
    const float rs = rsqrtf(var + CLN_EPS);

    float4 ov;
    union { __bf16 p[4]; uint2 u; } ob;
    #pragma unroll
    for (int j = 0; j < 4; ++j) {
        const int c = tid * 4 + j;
        const float o = (v[j] - mu) * rs * g[c] + bta[c];
        (&ov.x)[j] = o;
        ob.p[j] = (__bf16)o;
    }
    *reinterpret_cast<float4*>(x + row * CD + tid * 4) = ov;
    *reinterpret_cast<uint2*>(xb + row * CD + tid * 4) = ob.u;
}

// mask_col = cumsum(softmax(w_s))  — tiny (S=128), one block.
__global__ __launch_bounds__(128) void mask_kernel(
    const float* __restrict__ ws, float* __restrict__ mc)
{
    __shared__ float e[CS];
    const int t = threadIdx.x;
    e[t] = ws[t];
    __syncthreads();
    if (t == 0) {
        float m = e[0];
        for (int i = 1; i < CS; ++i) m = fmaxf(m, e[i]);
        float s = 0.f;
        for (int i = 0; i < CS; ++i) { e[i] = __expf(e[i] - m); s += e[i]; }
        float c = 0.f;
        const float inv = 1.0f / s;
        for (int i = 0; i < CS; ++i) { c += e[i] * inv; e[i] = c; }
    }
    __syncthreads();
    mc[t] = e[t];
}

__global__ __launch_bounds__(256) void f32_to_bf16(
    const float* __restrict__ s, __bf16* __restrict__ d, long n)
{
    long i = (long)blockIdx.x * blockDim.x + threadIdx.x;
    const long stride = (long)gridDim.x * blockDim.x;
    for (; i < n; i += stride) d[i] = (__bf16)s[i];
}

// Per-layer transpose+convert: wt[l][n][k] = (bf16) w[l][k][n].
__global__ __launch_bounds__(256) void w_transpose_bf16(
    const float* __restrict__ w, __bf16* __restrict__ wt, int K, int N, long total)
{
    long i = (long)blockIdx.x * blockDim.x + threadIdx.x;
    const long stride = (long)gridDim.x * blockDim.x;
    const long slab = (long)K * N;
    for (; i < total; i += stride) {
        const long l = i / slab, r = i % slab;
        const long n = r / K, k = r % K;
        wt[i] = (__bf16)w[l * slab + k * N + n];
    }
}

// ---------------------------------------------------------------------------
// Orchestration
// ---------------------------------------------------------------------------
extern "C" void kernel_launch(void* const* d_in, const int* in_sizes, int n_in,
                              void* d_out, int out_size, void* d_ws, size_t ws_size,
                              hipStream_t stream)
{
    (void)in_sizes; (void)n_in; (void)out_size; (void)ws_size;

    const long MD = (long)CB * CS * CD;       // 16.8M  (x / q / k / v / t)
    const long MF = (long)CB * CS * CF;       // 67.1M  (FFN hidden)
    const long MS = (long)CB * CH * CS * CS;  // 33.6M  (scores)

    const float* src  = (const float*)d_in[0];
    const float* w_s  = (const float*)d_in[1];
    const float* Wq   = (const float*)d_in[2];
    const float* bq   = (const float*)d_in[3];
    const float* Wk   = (const float*)d_in[4];
    const float* bk   = (const float*)d_in[5];
    const float* Wv   = (const float*)d_in[6];
    const float* bv   = (const float*)d_in[7];
    const float* Wo   = (const float*)d_in[8];
    const float* bo   = (const float*)d_in[9];
    const float* ln1g = (const float*)d_in[10];
    const float* ln1b = (const float*)d_in[11];
    const float* W1   = (const float*)d_in[12];
    const float* b1   = (const float*)d_in[13];
    const float* W2   = (const float*)d_in[14];
    const float* b2   = (const float*)d_in[15];
    const float* ln2g = (const float*)d_in[16];
    const float* ln2b = (const float*)d_in[17];

    char* ws = (char*)d_ws;
    size_t off = 0;
    auto take = [&](size_t bytes) -> char* {
        char* p = ws + off;
        off = (off + bytes + 255) & ~(size_t)255;
        return p;
    };

    float*  x     = (float*)take(MD * 4);
    __bf16* x_bf  = (__bf16*)take(MD * 2);
    float*  tacc  = (float*)take(MD * 4);
    __bf16* q_bf  = (__bf16*)take(MD * 2);
    __bf16* k_bf  = (__bf16*)take(MD * 2);
    __bf16* v_bf  = (__bf16*)take(MD * 2);   // [b,h,dh,s] layout (mode 3)
    __bf16* ao_bf = (__bf16*)take(MD * 2);
    __bf16* h_bf  = (__bf16*)take(MF * 2);
    float*  prev  = (float*)take(MS * 4);
    __bf16* p_bf  = (__bf16*)take(MS * 2);
    float*  maskc = (float*)take(CS * 4);
    __bf16* wq_t  = (__bf16*)take((size_t)CNL * CD * CD * 2);
    __bf16* wk_t  = (__bf16*)take((size_t)CNL * CD * CD * 2);
    __bf16* wv_t  = (__bf16*)take((size_t)CNL * CD * CD * 2);
    __bf16* wo_t  = (__bf16*)take((size_t)CNL * CD * CD * 2);
    __bf16* w1_t  = (__bf16*)take((size_t)CNL * CD * CF * 2);
    __bf16* w2_t  = (__bf16*)take((size_t)CNL * CF * CD * 2);

    // One-time prep (re-done every call: deterministic, capture-safe).
    const long nDD = (long)CNL * CD * CD, nDF = (long)CNL * CD * CF;
    w_transpose_bf16<<<1024, 256, 0, stream>>>(Wq, wq_t, CD, CD, nDD);
    w_transpose_bf16<<<1024, 256, 0, stream>>>(Wk, wk_t, CD, CD, nDD);
    w_transpose_bf16<<<1024, 256, 0, stream>>>(Wv, wv_t, CD, CD, nDD);
    w_transpose_bf16<<<1024, 256, 0, stream>>>(Wo, wo_t, CD, CD, nDD);
    w_transpose_bf16<<<2048, 256, 0, stream>>>(W1, w1_t, CD, CF, nDF);  // [F][D]
    w_transpose_bf16<<<2048, 256, 0, stream>>>(W2, w2_t, CF, CD, nDF);  // [D][F]
    hipMemcpyAsync(x, src, MD * 4, hipMemcpyDeviceToDevice, stream);
    f32_to_bf16<<<2048, 256, 0, stream>>>(src, x_bf, MD);
    mask_kernel<<<1, 128, 0, stream>>>(w_s, maskc);

    const int M = CB * CS;                 // 32768 rows
    const dim3 blk(256);
    const dim3 gD(CD / 128, M / 128);      // (4, 256)  for N=512 GEMMs
    const dim3 gF(CF / 128, M / 128);      // (16, 256) for N=2048 GEMM
    const int BH = CB * CH;                // 2048

    for (int i = 0; i < CNL; ++i) {
        const long wD = (long)i * CD * CD;
        const long wF = (long)i * CD * CF;

        // QKV projections (bf16 out; V written in [b,h,dh,s] layout)
        gemm_bf16<<<gD, blk, 0, stream>>>(x_bf, wq_t + wD, bq + (long)i*CD,
                                          nullptr, q_bf, M, CD, CD, 1);
        gemm_bf16<<<gD, blk, 0, stream>>>(x_bf, wk_t + wD, bk + (long)i*CD,
                                          nullptr, k_bf, M, CD, CD, 1);
        gemm_bf16<<<gD, blk, 0, stream>>>(x_bf, wv_t + wD, bv + (long)i*CD,
                                          nullptr, v_bf, M, CD, CD, 3);

        // attention: scores (+mask*scale +prev, stored back to prev), softmax, @V
        attn_scores <<<BH, blk, 0, stream>>>(q_bf, k_bf, maskc, prev, i > 0 ? 1 : 0);
        attn_softmax<<<(BH * CS) / 8, blk, 0, stream>>>(prev, p_bf);
        attn_v      <<<BH, blk, 0, stream>>>(p_bf, v_bf, ao_bf);

        // output projection + residual layernorm 1
        gemm_bf16<<<gD, blk, 0, stream>>>(ao_bf, wo_t + wD, bo + (long)i*CD,
                                          tacc, nullptr, M, CD, CD, 0);
        resid_ln<<<M, 128, 0, stream>>>(x, tacc, ln1g + (long)i*CD,
                                        ln1b + (long)i*CD, x_bf);

        // FFN: GELU(x @ W1 + b1) @ W2 + b2, residual layernorm 2
        gemm_bf16<<<gF, blk, 0, stream>>>(x_bf, w1_t + wF, b1 + (long)i*CF,
                                          nullptr, h_bf, M, CF, CD, 2);
        gemm_bf16<<<gD, blk, 0, stream>>>(h_bf, w2_t + wF, b2 + (long)i*CD,
                                          tacc, nullptr, M, CD, CF, 0);
        resid_ln<<<M, 128, 0, stream>>>(x, tacc, ln2g + (long)i*CD,
                                        ln2b + (long)i*CD, x_bf);
    }

    hipMemcpyAsync(d_out, x, MD * 4, hipMemcpyDeviceToDevice, stream);
}
